// BLTLocalEncoder_13348758356336
// MI455X (gfx1250) — matrix operand order repaired and checked by reading, not compiled, not run on hardware
//
#include <hip/hip_runtime.h>
#include <hip/hip_bf16.h>
#include <math.h>
#include <stdint.h>

// ---------------- problem dims (compile-time) ----------------
constexpr int Bq  = 4;
constexpr int Sq  = 2048;
constexpr int Dq  = 1024;
constexpr int Hq  = 16;
constexpr int DHq = 64;       // Dq / Hq
constexpr int Fq  = 4096;
constexpr int Lq  = 2;
constexpr float EPSq = 1e-5f;

typedef __bf16 bf16_t;
typedef __attribute__((ext_vector_type(16))) __bf16 v16bf;
typedef __attribute__((ext_vector_type(8)))  __bf16 v8bf;
typedef __attribute__((ext_vector_type(4)))  __bf16 v4bf;
typedef __attribute__((ext_vector_type(8)))  float  v8f;

static __device__ inline v8f zero_v8f() {
  v8f z = {0.f,0.f,0.f,0.f,0.f,0.f,0.f,0.f};
  return z;
}

static __device__ inline float gelu_f(float x) {
  return 0.5f * x * (1.0f + erff(x * 0.7071067811865476f));
}

// ---------------- CDNA5 async global->LDS copy (ASYNCcnt-tracked) ----------------
static __device__ inline void async_b128(void* lds, const void* gptr) {
  uint32_t l = (uint32_t)(uintptr_t)lds;   // LDS aperture: addr[31:0] = LDS offset
  asm volatile("global_load_async_to_lds_b128 %0, %1, off"
               :: "v"(l), "v"(gptr) : "memory");
}
#define WAIT_ASYNC_ALL()  asm volatile("s_wait_asynccnt 0x0" ::: "memory")
#define WAIT_ASYNC_6()    asm volatile("s_wait_asynccnt 0x6" ::: "memory")

// ---------------- WMMA fragment loaders from LDS ----------------
// A-matrix 16x32 bf16 fragment:
//   lanes 0-15 : row M = lane,    K = {0..7, 16..23}
//   lanes 16-31: row M = lane-16, K = {8..15, 24..31}
static __device__ inline v16bf load_frag_A(const bf16_t* base, int lane, int stride) {
  int row = lane & 15;
  int k0  = (lane >> 4) * 8;
  const bf16_t* p = base + row * stride + k0;
  v8bf lo = *(const v8bf*)(p);
  v8bf hi = *(const v8bf*)(p + 16);
  return __builtin_shufflevector(lo, hi, 0,1,2,3,4,5,6,7,8,9,10,11,12,13,14,15);
}

// B-matrix 32x16 bf16 fragment (LDS tile stored [n][k]):
//   lanes 0-15 : col N = lane,    K = {0..7, 16..23}
//   lanes 16-31: col N = lane-16, K = {8..15, 24..31}
static __device__ inline v16bf load_frag_B(const bf16_t* base, int lane, int stride) {
  int col = lane & 15;
  int k0  = (lane >> 4) * 8;
  const bf16_t* p = base + col * stride + k0;
  v8bf lo = *(const v8bf*)(p);
  v8bf hi = *(const v8bf*)(p + 16);
  return __builtin_shufflevector(lo, hi, 0,1,2,3,4,5,6,7,8,9,10,11,12,13,14,15);
}

static __device__ inline v8f wmma_bf16(v16bf a, v16bf b, v8f c) {
  return __builtin_amdgcn_wmma_f32_16x16x32_bf16(
      false, a, false, b, (short)0, c, false, false);
}

// ---------------- weight transpose + bf16 convert:  Wt[n][k] = (bf16)W[k][n] ----------------
__launch_bounds__(256)
__global__ void transpose_cvt_kernel(const float* __restrict__ W,
                                     bf16_t* __restrict__ Wt,
                                     int K, int N) {
  __shared__ float tile[32][33];
  int n0 = blockIdx.x * 32;
  int k0 = blockIdx.y * 32;
  int tx = threadIdx.x;        // 32
  int ty = threadIdx.y;        // 8
#pragma unroll
  for (int r = 0; r < 4; ++r)
    tile[ty + 8 * r][tx] = W[(size_t)(k0 + ty + 8 * r) * N + n0 + tx];
  __syncthreads();
#pragma unroll
  for (int r = 0; r < 4; ++r)
    Wt[(size_t)(n0 + ty + 8 * r) * K + k0 + tx] = (bf16_t)tile[tx][ty + 8 * r];
}

// ---------------- embedding + LN0 (fp32 out: residual stream) ----------------
__launch_bounds__(256)
__global__ void embed_ln_kernel(const int* __restrict__ ids,
                                const float* __restrict__ hash,
                                const float* __restrict__ tok,
                                const float* __restrict__ pos,
                                const float* __restrict__ g,
                                const float* __restrict__ bta,
                                float* __restrict__ Y) {
  int row = blockIdx.x;
  int s   = row & (Sq - 1);
  int tid = threadIdx.x;
  int id  = ids[row];
  float4 tv = ((const float4*)(tok  + (size_t)id  * Dq))[tid];
  float4 pv = ((const float4*)(pos  + (size_t)s   * Dq))[tid];
  float4 hv = ((const float4*)(hash + (size_t)row * Dq))[tid];
  float v0 = tv.x + pv.x + hv.x, v1 = tv.y + pv.y + hv.y;
  float v2 = tv.z + pv.z + hv.z, v3 = tv.w + pv.w + hv.w;

  float s1 = v0 + v1 + v2 + v3;
  float s2 = v0*v0 + v1*v1 + v2*v2 + v3*v3;
  __shared__ float r1[8], r2[8];
  __shared__ float mu_s, rstd_s;
  int lane = tid & 31, w = tid >> 5;
#pragma unroll
  for (int off = 16; off > 0; off >>= 1) {
    s1 += __shfl_down(s1, off, 32);
    s2 += __shfl_down(s2, off, 32);
  }
  if (lane == 0) { r1[w] = s1; r2[w] = s2; }
  __syncthreads();
  if (tid == 0) {
    float a = 0.f, b2 = 0.f;
#pragma unroll
    for (int i = 0; i < 8; ++i) { a += r1[i]; b2 += r2[i]; }
    float mu  = a / (float)Dq;
    float var = b2 / (float)Dq - mu * mu;
    mu_s = mu; rstd_s = rsqrtf(var + EPSq);
  }
  __syncthreads();
  float mu = mu_s, rs = rstd_s;
  float4 gv = ((const float4*)g)[tid];
  float4 bv = ((const float4*)bta)[tid];
  float4 o;
  o.x = (v0 - mu) * rs * gv.x + bv.x;
  o.y = (v1 - mu) * rs * gv.y + bv.y;
  o.z = (v2 - mu) * rs * gv.z + bv.z;
  o.w = (v3 - mu) * rs * gv.w + bv.w;
  ((float4*)(Y + (size_t)row * Dq))[tid] = o;
}

// ---------------- LayerNorm (fp32 in -> bf16 out, GEMM-ready) ----------------
__launch_bounds__(256)
__global__ void ln_bf16_kernel(const float* __restrict__ X,
                               const float* __restrict__ g,
                               const float* __restrict__ bta,
                               bf16_t* __restrict__ Y) {
  int row = blockIdx.x;
  int tid = threadIdx.x;
  float4 xv = ((const float4*)(X + (size_t)row * Dq))[tid];
  float s1 = xv.x + xv.y + xv.z + xv.w;
  float s2 = xv.x*xv.x + xv.y*xv.y + xv.z*xv.z + xv.w*xv.w;
  __shared__ float r1[8], r2[8];
  __shared__ float mu_s, rstd_s;
  int lane = tid & 31, w = tid >> 5;
#pragma unroll
  for (int off = 16; off > 0; off >>= 1) {
    s1 += __shfl_down(s1, off, 32);
    s2 += __shfl_down(s2, off, 32);
  }
  if (lane == 0) { r1[w] = s1; r2[w] = s2; }
  __syncthreads();
  if (tid == 0) {
    float a = 0.f, b2 = 0.f;
#pragma unroll
    for (int i = 0; i < 8; ++i) { a += r1[i]; b2 += r2[i]; }
    float mu  = a / (float)Dq;
    float var = b2 / (float)Dq - mu * mu;
    mu_s = mu; rstd_s = rsqrtf(var + EPSq);
  }
  __syncthreads();
  float mu = mu_s, rs = rstd_s;
  float4 gv = ((const float4*)g)[tid];
  float4 bv = ((const float4*)bta)[tid];
  v4bf o;
  o[0] = (bf16_t)((xv.x - mu) * rs * gv.x + bv.x);
  o[1] = (bf16_t)((xv.y - mu) * rs * gv.y + bv.y);
  o[2] = (bf16_t)((xv.z - mu) * rs * gv.z + bv.z);
  o[3] = (bf16_t)((xv.w - mu) * rs * gv.w + bv.w);
  ((v4bf*)(Y + (size_t)row * Dq))[tid] = o;
}

// ---------------- bf16 WMMA GEMM, async double-buffered LDS, BK=64 ----------------
// C = act(A @ Wt^T + bias) (+ Res).  A: [M,K] bf16, Wt: [N,K] bf16 (pre-transposed),
// bias/Res fp32.  ACT: 0 none, 1 gelu, 2 scale by 1/sqrt(DH).
// Block tile 128x64, BK=64 (8 WMMAs per staged tile), 256 threads, wave tile 32x32.
template<int ACT, bool RES, typename OutT>
static __device__ inline void store_acc(const v8f& acc, int row0, int col0, int N,
                                        const float* __restrict__ bias,
                                        const float* __restrict__ Res,
                                        OutT* __restrict__ C,
                                        int halfoff, int nn) {
#pragma unroll
  for (int j = 0; j < 8; ++j) {
    int row = row0 + j + halfoff;
    int col = col0 + nn;
    float v = acc[j] + bias[col];
    if (ACT == 1) v = gelu_f(v);
    if (ACT == 2) v *= 0.125f;
    if (RES) v += Res[(size_t)row * N + col];
    C[(size_t)row * N + col] = (OutT)v;
  }
}

template<typename OutT, int ACT, bool RES>
__launch_bounds__(256)
__global__ void gemm_bf16_kernel(const bf16_t* __restrict__ A,
                                 const bf16_t* __restrict__ Wt,   // [N][K]
                                 const float*  __restrict__ bias,
                                 const float*  __restrict__ Res,
                                 OutT* __restrict__ C,
                                 int M, int N, int K) {
  __shared__ bf16_t As[2][128][72];   // [m][k], 144B rows keep b128 alignment
  __shared__ bf16_t Ws[2][64][72];    // [n][k]

  int tid  = threadIdx.x;
  int lane = tid & 31;
  int w    = tid >> 5;
  int wr   = (w & 3) * 32;
  int wc   = (w >> 2) * 32;
  int rowBase = blockIdx.x * 128;
  int colBase = blockIdx.y * 64;

  // A tile: 128 rows x 64 elems = 1024 x 16B chunks -> 4 per thread
  // W tile:  64 rows x 64 elems =  512 chunks       -> 2 per thread
  int arow[4], aelm[4], wrow2[2], welm2[2];
#pragma unroll
  for (int p = 0; p < 4; ++p) {
    int c = tid + p * 256;
    arow[p] = c >> 3;
    aelm[p] = (c & 7) * 8;
  }
#pragma unroll
  for (int p = 0; p < 2; ++p) {
    int c = tid + p * 256;
    wrow2[p] = c >> 3;
    welm2[p] = (c & 7) * 8;
  }

  v8f c00 = zero_v8f(), c01 = zero_v8f(), c10 = zero_v8f(), c11 = zero_v8f();

  const int nk = K >> 6;
  // prologue: stage tile 0 (6 async instructions per wave)
#pragma unroll
  for (int p = 0; p < 4; ++p)
    async_b128(&As[0][arow[p]][aelm[p]],
               A + (size_t)(rowBase + arow[p]) * K + aelm[p]);
#pragma unroll
  for (int p = 0; p < 2; ++p)
    async_b128(&Ws[0][wrow2[p]][welm2[p]],
               Wt + (size_t)(colBase + wrow2[p]) * K + welm2[p]);

  for (int i = 0; i < nk; ++i) {
    int cur = i & 1;
    if (i + 1 < nk) {
      int kt = (i + 1) << 6;
#pragma unroll
      for (int p = 0; p < 4; ++p)
        async_b128(&As[cur ^ 1][arow[p]][aelm[p]],
                   A + (size_t)(rowBase + arow[p]) * K + kt + aelm[p]);
#pragma unroll
      for (int p = 0; p < 2; ++p)
        async_b128(&Ws[cur ^ 1][wrow2[p]][welm2[p]],
                   Wt + (size_t)(colBase + wrow2[p]) * K + kt + welm2[p]);
      WAIT_ASYNC_6();          // current tile's batch retired, next in flight
    } else {
      WAIT_ASYNC_ALL();
    }
    __syncthreads();

#pragma unroll
    for (int kk = 0; kk < 64; kk += 32) {
      v16bf a0 = load_frag_A(&As[cur][wr][kk],      lane, 72);
      v16bf a1 = load_frag_A(&As[cur][wr + 16][kk], lane, 72);
      v16bf b0 = load_frag_B(&Ws[cur][wc][kk],      lane, 72);
      v16bf b1 = load_frag_B(&Ws[cur][wc + 16][kk], lane, 72);
      c00 = wmma_bf16(a0, b0, c00);
      c01 = wmma_bf16(a0, b1, c01);
      c10 = wmma_bf16(a1, b0, c10);
      c11 = wmma_bf16(a1, b1, c11);
    }
    __syncthreads();           // reads done before this buffer is refilled
  }

  int halfoff = (lane >> 4) * 8;
  int nn = lane & 15;
  store_acc<ACT, RES, OutT>(c00, rowBase + wr,      colBase + wc,      N, bias, Res, C, halfoff, nn);
  store_acc<ACT, RES, OutT>(c01, rowBase + wr,      colBase + wc + 16, N, bias, Res, C, halfoff, nn);
  store_acc<ACT, RES, OutT>(c10, rowBase + wr + 16, colBase + wc,      N, bias, Res, C, halfoff, nn);
  store_acc<ACT, RES, OutT>(c11, rowBase + wr + 16, colBase + wc + 16, N, bias, Res, C, halfoff, nn);
}

// ---------------- flash attention with patch-equality bias (bf16 q/k/v/ctx) ----------------
// grid: (S/64, B*H). block: 256 threads (8 waves). wave tile: 16 rows x 32 cols.
// 1/sqrt(DH) already folded into Q by the projection epilogue (ACT=2).
__launch_bounds__(256)
__global__ void attn_kernel(const bf16_t* __restrict__ Q,
                            const bf16_t* __restrict__ Kd,
                            const bf16_t* __restrict__ V,
                            const int*    __restrict__ patch,
                            bf16_t* __restrict__ Ctx) {
  __shared__ bf16_t q_s[64][72];   // [query][dh]
  __shared__ bf16_t k_s[64][72];   // [key][dh]
  __shared__ bf16_t v_s[64][72];   // [dh][key]  (transposed V)
  __shared__ bf16_t p_s[64][72];   // probs bf16
  __shared__ float  s_s[64][68];   // raw scores fp32
  __shared__ float  m_s[64], l_s[64], r_s[64];
  __shared__ int    pq_s[64], pk_s[64];

  int qb = blockIdx.x;
  int bh = blockIdx.y;
  int b  = bh / Hq;
  int h  = bh - b * Hq;
  int tid = threadIdx.x, lane = tid & 31, w = tid >> 5;
  int wr = (w & 3) * 16;
  int wc = (w >> 2) * 32;
  int s0 = qb * 64;

  // q tile: 64 rows x 64 elems = 512 x 16B chunks -> 2 async per thread
  int qr0 = tid >> 3,         qe0 = (tid & 7) * 8;
  int qr1 = (tid + 256) >> 3, qe1 = (tid & 7) * 8;
  async_b128(&q_s[qr0][qe0], Q + ((size_t)b * Sq + s0 + qr0) * Dq + h * DHq + qe0);
  async_b128(&q_s[qr1][qe1], Q + ((size_t)b * Sq + s0 + qr1) * Dq + h * DHq + qe1);

  if (tid < 64) {
    pq_s[tid] = patch[(size_t)b * Sq + s0 + tid];
    m_s[tid]  = -3.0e38f;
    l_s[tid]  = 0.f;
  }

  v8f o0 = zero_v8f(), o1 = zero_v8f();
  int halfoff = (lane >> 4) * 8;
  int nn = lane & 15;

  for (int kb = 0; kb < Sq / 64; ++kb) {
    int k0r = kb * 64;
    __syncthreads();   // previous iteration's LDS reads complete
    // K tile via async copy
    async_b128(&k_s[qr0][qe0], Kd + ((size_t)b * Sq + k0r + qr0) * Dq + h * DHq + qe0);
    async_b128(&k_s[qr1][qe1], Kd + ((size_t)b * Sq + k0r + qr1) * Dq + h * DHq + qe1);
    // V tile staged manually (needs transpose)
    {
      int row = tid >> 2;
      int c0  = (tid & 3) * 16;
      const v8bf* vp = (const v8bf*)(V + ((size_t)b * Sq + k0r + row) * Dq + h * DHq + c0);
      v8bf u0 = vp[0], u1 = vp[1];
#pragma unroll
      for (int i = 0; i < 8; ++i) {
        v_s[c0 + i][row]     = u0[i];
        v_s[c0 + 8 + i][row] = u1[i];
      }
    }
    if (tid < 64) pk_s[tid] = patch[(size_t)b * Sq + k0r + tid];
    WAIT_ASYNC_ALL();
    __syncthreads();

    // scores = q @ k^T  (K-dim = DH = 64 -> two wmma k-steps)
    v8f sc0 = zero_v8f(), sc1 = zero_v8f();
#pragma unroll
    for (int kk = 0; kk < 64; kk += 32) {
      v16bf aF  = load_frag_A(&q_s[wr][kk],      lane, 72);
      v16bf bF0 = load_frag_B(&k_s[wc][kk],      lane, 72);
      v16bf bF1 = load_frag_B(&k_s[wc + 16][kk], lane, 72);
      sc0 = wmma_bf16(aF, bF0, sc0);
      sc1 = wmma_bf16(aF, bF1, sc1);
    }
    // patch-equality bias, spill scores
#pragma unroll
    for (int j = 0; j < 8; ++j) {
      int m  = wr + j + halfoff;
      int pm = pq_s[m];
      float bia0 = (pm == pk_s[wc + nn])      ? 0.f : -10000.f;
      float bia1 = (pm == pk_s[wc + 16 + nn]) ? 0.f : -10000.f;
      s_s[m][wc + nn]      = sc0[j] + bia0;
      s_s[m][wc + 16 + nn] = sc1[j] + bia1;
    }
    __syncthreads();

    // online softmax (threads 0..63 own a row each)
    if (tid < 64) {
      float mo = m_s[tid];
      float mx = mo;
#pragma unroll 8
      for (int j2 = 0; j2 < 64; ++j2) mx = fmaxf(mx, s_s[tid][j2]);
      float rsc = __expf(mo - mx);
      float sum = 0.f;
#pragma unroll 8
      for (int j2 = 0; j2 < 64; ++j2) {
        float pv = __expf(s_s[tid][j2] - mx);
        p_s[tid][j2] = (bf16_t)pv;
        sum += pv;
      }
      m_s[tid] = mx;
      l_s[tid] = l_s[tid] * rsc + sum;
      r_s[tid] = rsc;
    }
    __syncthreads();

    // rescale running output, then O += P @ V
#pragma unroll
    for (int j = 0; j < 8; ++j) {
      float rr = r_s[wr + j + halfoff];
      o0[j] *= rr;
      o1[j] *= rr;
    }
#pragma unroll
    for (int kk = 0; kk < 64; kk += 32) {
      v16bf aF  = load_frag_A(&p_s[wr][kk],      lane, 72);
      v16bf bF0 = load_frag_B(&v_s[wc][kk],      lane, 72);
      v16bf bF1 = load_frag_B(&v_s[wc + 16][kk], lane, 72);
      o0 = wmma_bf16(aF, bF0, o0);
      o1 = wmma_bf16(aF, bF1, o1);
    }
  }

  // finalize: divide by l, write ctx bf16
#pragma unroll
  for (int j = 0; j < 8; ++j) {
    int m = wr + j + halfoff;
    float linv = 1.0f / l_s[m];
    size_t base = ((size_t)b * Sq + (s0 + m)) * Dq + h * DHq;
    Ctx[base + wc + nn]      = (bf16_t)(o0[j] * linv);
    Ctx[base + wc + 16 + nn] = (bf16_t)(o1[j] * linv);
  }
}

// ---------------- host orchestration ----------------
extern "C" void kernel_launch(void* const* d_in, const int* in_sizes, int n_in,
                              void* d_out, int out_size, void* d_ws, size_t ws_size,
                              hipStream_t stream) {
  (void)in_sizes; (void)n_in; (void)out_size; (void)ws_size;

  const int*   input_ids = (const int*)d_in[0];
  const int*   patch_ids = (const int*)d_in[1];
  const float* hash_emb  = (const float*)d_in[2];
  const float* tok_emb   = (const float*)d_in[3];
  const float* pos_emb   = (const float*)d_in[4];
  const float* ln0_g = (const float*)d_in[5];
  const float* ln0_b = (const float*)d_in[6];
  const float* ln1_g = (const float*)d_in[7];
  const float* ln1_b = (const float*)d_in[8];
  const float* Wq = (const float*)d_in[9];
  const float* bq = (const float*)d_in[10];
  const float* Wk = (const float*)d_in[11];
  const float* bk = (const float*)d_in[12];
  const float* Wv = (const float*)d_in[13];
  const float* bv = (const float*)d_in[14];
  const float* Wo = (const float*)d_in[15];
  const float* bo = (const float*)d_in[16];
  const float* ln2_g = (const float*)d_in[17];
  const float* ln2_b = (const float*)d_in[18];
  const float* W1 = (const float*)d_in[19];
  const float* b1 = (const float*)d_in[20];
  const float* W2 = (const float*)d_in[21];
  const float* b2 = (const float*)d_in[22];

  const int ROWS = Bq * Sq;                 // 8192
  const size_t nBSD = (size_t)ROWS * Dq;    // 8.39M elems

  // workspace layout
  float*  Hb  = (float*)d_ws;               // residual stream fp32
  bf16_t* Xb  = (bf16_t*)(Hb + nBSD);       // LN output bf16
  bf16_t* Qb  = Xb + nBSD;
  bf16_t* Kb2 = Qb + nBSD;
  bf16_t* Vb2 = Kb2 + nBSD;
  bf16_t* Cx  = Vb2 + nBSD;                 // attention ctx bf16
  bf16_t* Tb  = Qb;                         // FFN hidden [ROWS,F] bf16: spans Qb..Cx exactly
  bf16_t* Wts = Cx + nBSD;                  // transposed bf16 weights
  const size_t DD = (size_t)Dq * Dq;
  const size_t DF = (size_t)Dq * Fq;
  bf16_t* WtQ = Wts;                        // [L][D][D]
  bf16_t* WtK = WtQ + Lq * DD;
  bf16_t* WtV = WtK + Lq * DD;
  bf16_t* WtO = WtV + Lq * DD;
  bf16_t* Wt1 = WtO + Lq * DD;              // [L][F][D]
  bf16_t* Wt2 = Wt1 + Lq * DF;              // [L][D][F]

  dim3 blk(256);
  dim3 tb(32, 8);
  dim3 gTdd(Dq / 32, Dq / 32);
  dim3 gTdf(Fq / 32, Dq / 32);              // W1: K=D, N=F
  dim3 gTfd(Dq / 32, Fq / 32);              // W2: K=F, N=D
  dim3 gD(ROWS / 128, Dq / 64);
  dim3 gF(ROWS / 128, Fq / 64);
  dim3 gA(Sq / 64, Bq * Hq);

  // one-time weight transpose + bf16 conversion
  for (int l = 0; l < Lq; ++l) {
    transpose_cvt_kernel<<<gTdd, tb, 0, stream>>>(Wq + l * DD, WtQ + l * DD, Dq, Dq);
    transpose_cvt_kernel<<<gTdd, tb, 0, stream>>>(Wk + l * DD, WtK + l * DD, Dq, Dq);
    transpose_cvt_kernel<<<gTdd, tb, 0, stream>>>(Wv + l * DD, WtV + l * DD, Dq, Dq);
    transpose_cvt_kernel<<<gTdd, tb, 0, stream>>>(Wo + l * DD, WtO + l * DD, Dq, Dq);
    transpose_cvt_kernel<<<gTdf, tb, 0, stream>>>(W1 + l * DF, Wt1 + l * DF, Dq, Fq);
    transpose_cvt_kernel<<<gTfd, tb, 0, stream>>>(W2 + l * DF, Wt2 + l * DF, Fq, Dq);
  }

  // h = LN0(tok[ids] + pos + hash)
  embed_ln_kernel<<<ROWS, blk, 0, stream>>>(input_ids, hash_emb, tok_emb, pos_emb,
                                            ln0_g, ln0_b, Hb);

  for (int l = 0; l < Lq; ++l) {
    // x = LN1(h)  (bf16)
    ln_bf16_kernel<<<ROWS, blk, 0, stream>>>(Hb, ln1_g + l * Dq, ln1_b + l * Dq, Xb);

    // q (scaled by 1/8 in epilogue), k, v projections -> bf16
    gemm_bf16_kernel<bf16_t, 2, false><<<gD, blk, 0, stream>>>(
        Xb, WtQ + l * DD, bq + l * Dq, nullptr, Qb, ROWS, Dq, Dq);
    gemm_bf16_kernel<bf16_t, 0, false><<<gD, blk, 0, stream>>>(
        Xb, WtK + l * DD, bk + l * Dq, nullptr, Kb2, ROWS, Dq, Dq);
    gemm_bf16_kernel<bf16_t, 0, false><<<gD, blk, 0, stream>>>(
        Xb, WtV + l * DD, bv + l * Dq, nullptr, Vb2, ROWS, Dq, Dq);

    // ctx = softmax(q k^T + patch bias) v  -> bf16
    attn_kernel<<<gA, blk, 0, stream>>>(Qb, Kb2, Vb2, patch_ids, Cx);

    // h = h + ctx @ Wo + bo   (fp32 residual stream)
    gemm_bf16_kernel<float, 0, true><<<gD, blk, 0, stream>>>(
        Cx, WtO + l * DD, bo + l * Dq, Hb, Hb, ROWS, Dq, Dq);

    // x = LN2(h)
    ln_bf16_kernel<<<ROWS, blk, 0, stream>>>(Hb, ln2_g + l * Dq, ln2_b + l * Dq, Xb);

    // t = gelu(x @ W1 + b1)  (bf16)
    gemm_bf16_kernel<bf16_t, 1, false><<<gF, blk, 0, stream>>>(
        Xb, Wt1 + l * DF, b1 + (size_t)l * Fq, nullptr, Tb, ROWS, Fq, Dq);

    // h = h + t @ W2 + b2  (last layer writes straight to d_out)
    float* outC = (l == Lq - 1) ? (float*)d_out : Hb;
    gemm_bf16_kernel<float, 0, true><<<gD, blk, 0, stream>>>(
        Tb, Wt2 + l * DF, b2 + l * Dq, Hb, outC, ROWS, Dq, Fq);
  }
}
